// ChunkedCrossAttention_85907935855128
// MI455X (gfx1250) — matrix operand order repaired and checked by reading, not compiled, not run on hardware
//
#include <hip/hip_runtime.h>
#include <stdint.h>

#define H_     8
#define DH_    64
#define CS_    64
#define DIM_   1024
#define INNER_ 512
#define NPOS_  4096
#define LKV_   256
#define LPAD_  288

typedef __attribute__((ext_vector_type(16))) __bf16 v16bf;
typedef __attribute__((ext_vector_type(8)))  float  v8f;

__device__ __forceinline__ uint32_t bf16b(float f) {
  uint32_t u = __float_as_uint(f);
  return ((u + 0x7FFFu + ((u >> 16) & 1u)) >> 16) & 0xFFFFu;  // RNE
}
__device__ __forceinline__ uint32_t bf16pk(float lo, float hi) {
  return bf16b(lo) | (bf16b(hi) << 16);
}
__device__ __forceinline__ float bf16f(uint32_t h) {
  return __uint_as_float((h & 0xFFFFu) << 16);
}

// 16-bit A-matrix fragment K offset: lanes<16 hold K{0..15}, lanes>=16 hold K{16..31}
__device__ __forceinline__ int a_koff(int jj, int khalf) {
  return 2 * jj + khalf * 16;
}

__device__ __forceinline__ v8f wmma_bf16(const uint32_t aw[8], const uint32_t bw[8], v8f c) {
  union { uint32_t w[8]; v16bf v; } A, B;
#pragma unroll
  for (int i = 0; i < 8; ++i) { A.w[i] = aw[i]; B.w[i] = bw[i]; }
  return __builtin_amdgcn_wmma_f32_16x16x32_bf16(false, A.v, false, B.v, (short)0, c, false, false);
}

// ---------------- Stage 1: QKV projections (f32 in -> bf16 out) ----------------
// MODE 0: A rows are the 63-shifted x (zero past n). MODE 1: A rows = context rows.
template <int MODE>
__global__ void gemm_proj(const float* __restrict__ Asrc, const float* __restrict__ B,
                          uint16_t* __restrict__ D, float alpha) {
  __shared__ uint32_t Alds[128 * 16];  // [row][kpair] bf16x2
  __shared__ uint32_t Blds[64 * 16];   // [n][kpair]  bf16x2
  int tid = threadIdx.x, lane = tid & 31, wave = tid >> 5;
  int khalf = lane >> 4, ln = lane & 15;
  int mbase = blockIdx.y * 128, nbase = blockIdx.x * 64;
  v8f acc[4];
#pragma unroll
  for (int t = 0; t < 4; ++t)
#pragma unroll
    for (int e = 0; e < 8; ++e) acc[t][e] = 0.f;

  for (int k0 = 0; k0 < DIM_; k0 += 32) {
#pragma unroll
    for (int it = 0; it < 8; ++it) {  // 128 rows x 16 kpairs
      int idx = it * 256 + tid, row = idx >> 4, kp = idx & 15;
      int g = mbase + row;
      float a0 = 0.f, a1 = 0.f;
      const float* rp = nullptr;
      if (MODE == 0) {
        int bkI = g >> 6, i = g & 63;
        int bq = bkI >> 6, ch = bkI & 63;
        int p = ch * 64 + i + 63;
        if (p < NPOS_) rp = Asrc + ((size_t)bq * NPOS_ + p) * DIM_;
      } else {
        rp = Asrc + (size_t)g * DIM_;
      }
      if (rp) { a0 = rp[k0 + kp * 2]; a1 = rp[k0 + kp * 2 + 1]; }
      Alds[row * 16 + kp] = bf16pk(a0, a1);
    }
#pragma unroll
    for (int it = 0; it < 4; ++it) {  // 64 n x 16 kpairs
      int idx = it * 256 + tid, n = idx & 63, kp = idx >> 6;
      int kk = k0 + kp * 2;
      Blds[n * 16 + kp] = bf16pk(B[(size_t)kk * INNER_ + nbase + n],
                                 B[(size_t)(kk + 1) * INNER_ + nbase + n]);
    }
    __syncthreads();
    uint32_t afr[8];
#pragma unroll
    for (int jj = 0; jj < 8; ++jj)
      afr[jj] = Alds[(wave * 16 + ln) * 16 + (a_koff(jj, khalf) >> 1)];
#pragma unroll
    for (int t = 0; t < 4; ++t) {
      uint32_t bfr[8];
#pragma unroll
      for (int jj = 0; jj < 8; ++jj)
        bfr[jj] = Blds[(t * 16 + ln) * 16 + ((khalf * 16 + 2 * jj) >> 1)];
      acc[t] = wmma_bf16(afr, bfr, acc[t]);
    }
    __syncthreads();
  }
#pragma unroll
  for (int t = 0; t < 4; ++t)
#pragma unroll
    for (int jj = 0; jj < 8; ++jj) {
      int m = mbase + wave * 16 + jj + 8 * khalf;
      int n = nbase + t * 16 + ln;
      D[(size_t)m * INNER_ + n] = (uint16_t)bf16b(acc[t][jj] * alpha);
    }
}

// ---------------- Rope (in place on bf16 buffers) ----------------
__device__ __forceinline__ void rope_row(uint16_t* row, const float* fr) {
  int t = threadIdx.x;
  int c0 = t, c1 = t + 256;
  int d0 = c0 & 63, d1 = c1 & 63;
  int pc0 = (d0 < 32) ? c0 + 32 : c0 - 32;
  int pc1 = (d1 < 32) ? c1 + 32 : c1 - 32;
  float v0 = bf16f(row[c0]), v1 = bf16f(row[c1]);
  float p0 = bf16f(row[pc0]), p1 = bf16f(row[pc1]);
  float f0 = fr[d0], f1 = fr[d1];
  __syncthreads();  // all reads before any write (rotate-half partners)
  float r0 = (d0 < 32) ? -p0 : p0;
  float r1 = (d1 < 32) ? -p1 : p1;
  row[c0] = (uint16_t)bf16b(v0 * __cosf(f0) + r0 * __sinf(f0));
  row[c1] = (uint16_t)bf16b(v1 * __cosf(f1) + r1 * __sinf(f1));
}

__global__ void rope_q_kernel(uint16_t* __restrict__ Q, const float* __restrict__ qpe) {
  // padded qpe: only row i==0 of each chunk has nonzero freqs (= q_pos_emb[63])
  uint16_t* row = Q + (size_t)blockIdx.x * CS_ * INNER_;
  rope_row(row, qpe + 63 * DH_);
}

__global__ void rope_k_kernel(uint16_t* __restrict__ K, const float* __restrict__ kpe) {
  int g = blockIdx.x;                       // row in (bk*256)
  uint16_t* row = K + (size_t)g * INNER_;
  rope_row(row, kpe + (size_t)(g & 127) * DH_);  // kpe tiled over r=2
}

// ---------------- Stage 2: attention per (bk, h) ----------------
__global__ void attn_kernel(const uint32_t* __restrict__ Qw, const uint32_t* __restrict__ Kw,
                            const uint16_t* __restrict__ V16, const float* __restrict__ nullk,
                            const float* __restrict__ nullv, uint16_t* __restrict__ O) {
  __shared__ uint32_t Kld[LPAD_ * 32];      // K' [j][dpair], j=0 null, 1..256 real, rest 0
  __shared__ uint32_t Vld[64 * 144];        // V' [d][jpair]
  __shared__ float    Pld[4][16 * LPAD_];   // per-wave sim/prob rows
  int h = blockIdx.x, bk = blockIdx.y;
  int tid = threadIdx.x, lane = tid & 31, wave = tid >> 5;
  int khalf = lane >> 4, ln = lane & 15;
  size_t kvBase = (size_t)bk * LKV_ * INNER_;

  for (int it = 0; it < 72; ++it) {         // 288*32 words
    int idx = it * 128 + tid;
    int j = idx >> 5, dp = idx & 31, d0 = dp * 2;
    uint32_t w;
    if (j == 0)            w = bf16pk(nullk[h * 64 + d0], nullk[h * 64 + d0 + 1]);
    else if (j <= LKV_)    w = Kw[((kvBase + (size_t)(j - 1) * INNER_) >> 1) + ((h * 64 + d0) >> 1)];
    else                   w = 0u;
    Kld[j * 32 + dp] = w;
  }
  for (int it = 0; it < 72; ++it) {         // 64*144 words
    int idx = it * 128 + tid;
    int d = idx / 144, jp = idx % 144;
    int j0 = jp * 2, j1 = jp * 2 + 1;
    uint32_t lo = (j0 == 0) ? bf16b(nullv[h * 64 + d])
                : (j0 <= LKV_) ? (uint32_t)V16[kvBase + (size_t)(j0 - 1) * INNER_ + h * 64 + d] : 0u;
    uint32_t hi = (j1 <= LKV_) ? (uint32_t)V16[kvBase + (size_t)(j1 - 1) * INNER_ + h * 64 + d] : 0u;
    Vld[d * 144 + jp] = lo | (hi << 16);
  }
  __syncthreads();

  // persistent Q fragments for this wave's 16 rows (d-steps 0 and 32)
  uint32_t qa[2][8];
  size_t qrow = (size_t)(bk * CS_ + wave * 16 + ln) * 256;  // 256 words per Q row
#pragma unroll
  for (int s = 0; s < 2; ++s)
#pragma unroll
    for (int jj = 0; jj < 8; ++jj)
      qa[s][jj] = Qw[qrow + ((h * 64 + s * 32 + a_koff(jj, khalf)) >> 1)];

  // sim = q . k'^T  -> Pld
  for (int jt = 0; jt < 18; ++jt) {
    v8f acc;
#pragma unroll
    for (int e = 0; e < 8; ++e) acc[e] = 0.f;
#pragma unroll
    for (int s = 0; s < 2; ++s) {
      uint32_t bfr[8];
#pragma unroll
      for (int jj = 0; jj < 8; ++jj)
        bfr[jj] = Kld[(jt * 16 + ln) * 32 + ((s * 32 + khalf * 16 + 2 * jj) >> 1)];
      acc = wmma_bf16(qa[s], bfr, acc);
    }
#pragma unroll
    for (int jj = 0; jj < 8; ++jj)
      Pld[wave][(jj + 8 * khalf) * LPAD_ + jt * 16 + ln] = acc[jj];
  }
  __syncthreads();

  // softmax over 257 valid columns, one thread per q row
  if (tid < 64) {
    float* pr = &Pld[tid >> 4][(tid & 15) * LPAD_];
    float mx = -3.0e38f;
    for (int c = 0; c <= LKV_; ++c) mx = fmaxf(mx, pr[c]);
    float sum = 0.f;
    for (int c = 0; c <= LKV_; ++c) { float e = __expf(pr[c] - mx); pr[c] = e; sum += e; }
    float inv = 1.f / sum;
    for (int c = 0; c <= LKV_; ++c) pr[c] *= inv;
    for (int c = LKV_ + 1; c < LPAD_; ++c) pr[c] = 0.f;
  }
  __syncthreads();

  // o = p . v'
  v8f acc2[4];
#pragma unroll
  for (int t = 0; t < 4; ++t)
#pragma unroll
    for (int e = 0; e < 8; ++e) acc2[t][e] = 0.f;
  for (int kj = 0; kj < LPAD_; kj += 32) {
    uint32_t afr[8];
#pragma unroll
    for (int jj = 0; jj < 8; ++jj) {
      int kb = kj + a_koff(jj, khalf);
      afr[jj] = bf16pk(Pld[wave][ln * LPAD_ + kb], Pld[wave][ln * LPAD_ + kb + 1]);
    }
#pragma unroll
    for (int t = 0; t < 4; ++t) {
      uint32_t bfr[8];
#pragma unroll
      for (int jj = 0; jj < 8; ++jj)
        bfr[jj] = Vld[(t * 16 + ln) * 144 + ((kj + khalf * 16 + 2 * jj) >> 1)];
      acc2[t] = wmma_bf16(afr, bfr, acc2[t]);
    }
  }
#pragma unroll
  for (int t = 0; t < 4; ++t)
#pragma unroll
    for (int jj = 0; jj < 8; ++jj) {
      int m = bk * CS_ + wave * 16 + jj + 8 * khalf;
      int col = h * 64 + t * 16 + ln;
      O[(size_t)m * INNER_ + col] = (uint16_t)bf16b(acc2[t][jj]);
    }
}

// ---------------- Stage 3: output projection + shifted f32 store ----------------
__global__ void gemm_out(const uint32_t* __restrict__ Ow, const float* __restrict__ Wo,
                         const float* __restrict__ bo, float* __restrict__ out) {
  __shared__ uint32_t Alds[128 * 16];
  __shared__ uint32_t Blds[64 * 16];
  int tid = threadIdx.x, lane = tid & 31, wave = tid >> 5;
  int khalf = lane >> 4, ln = lane & 15;
  int mbase = blockIdx.y * 128, nbase = blockIdx.x * 64;
  v8f acc[4];
#pragma unroll
  for (int t = 0; t < 4; ++t)
#pragma unroll
    for (int e = 0; e < 8; ++e) acc[t][e] = 0.f;

  for (int k0 = 0; k0 < INNER_; k0 += 32) {
#pragma unroll
    for (int it = 0; it < 8; ++it) {
      int idx = it * 256 + tid, row = idx >> 4, kp = idx & 15;
      Alds[row * 16 + kp] = Ow[(size_t)(mbase + row) * 256 + (k0 >> 1) + kp];
    }
#pragma unroll
    for (int it = 0; it < 4; ++it) {
      int idx = it * 256 + tid, n = idx & 63, kp = idx >> 6;
      int kk = k0 + kp * 2;
      Blds[n * 16 + kp] = bf16pk(Wo[(size_t)kk * DIM_ + nbase + n],
                                 Wo[(size_t)(kk + 1) * DIM_ + nbase + n]);
    }
    __syncthreads();
    uint32_t afr[8];
#pragma unroll
    for (int jj = 0; jj < 8; ++jj)
      afr[jj] = Alds[(wave * 16 + ln) * 16 + (a_koff(jj, khalf) >> 1)];
#pragma unroll
    for (int t = 0; t < 4; ++t) {
      uint32_t bfr[8];
#pragma unroll
      for (int jj = 0; jj < 8; ++jj)
        bfr[jj] = Blds[(t * 16 + ln) * 16 + ((khalf * 16 + 2 * jj) >> 1)];
      acc[t] = wmma_bf16(afr, bfr, acc[t]);
    }
    __syncthreads();
  }
#pragma unroll
  for (int t = 0; t < 4; ++t)
#pragma unroll
    for (int jj = 0; jj < 8; ++jj) {
      int g = mbase + wave * 16 + jj + 8 * khalf;
      int b = g >> 12, tpos = g & 4095;
      int n = nbase + t * 16 + ln;
      if (tpos < NPOS_ - 63)  // dest row = tpos + 63
        out[((size_t)b * NPOS_ + tpos + 63) * DIM_ + n] = acc[t][jj] + bo[n];
    }
}

__global__ void zero_prefix(float* __restrict__ out) {
  int idx = blockIdx.x * 256 + threadIdx.x;
  const int per = 63 * DIM_;
  if (idx < 4 * per) {
    int b = idx / per, r = idx % per;
    out[(size_t)b * NPOS_ * DIM_ + r] = 0.f;
  }
}

extern "C" void kernel_launch(void* const* d_in, const int* in_sizes, int n_in,
                              void* d_out, int out_size, void* d_ws, size_t ws_size,
                              hipStream_t stream) {
  (void)in_sizes; (void)n_in; (void)out_size; (void)ws_size;
  const float* x   = (const float*)d_in[0];
  const float* ctx = (const float*)d_in[1];
  const float* qpe = (const float*)d_in[2];
  const float* kpe = (const float*)d_in[3];
  const float* Wq  = (const float*)d_in[4];
  const float* Wk  = (const float*)d_in[5];
  const float* Wv  = (const float*)d_in[6];
  const float* Wo  = (const float*)d_in[7];
  const float* bo  = (const float*)d_in[8];
  const float* nk  = (const float*)d_in[9];
  const float* nv  = (const float*)d_in[10];
  float* out = (float*)d_out;

  char* ws = (char*)d_ws;
  const size_t QB = (size_t)16384 * 512 * 2;   // 16 MB
  const size_t KB = (size_t)65536 * 512 * 2;   // 64 MB
  uint16_t* Qb = (uint16_t*)(ws);
  uint16_t* Kb = (uint16_t*)(ws + QB);
  uint16_t* Vb = (uint16_t*)(ws + QB + KB);
  uint16_t* Ob = (uint16_t*)(ws + QB + 2 * KB);

  zero_prefix<<<(4 * 63 * DIM_ + 255) / 256, 256, 0, stream>>>(out);
  gemm_proj<0><<<dim3(8, 128), 256, 0, stream>>>(x,   Wq, Qb, 0.125f);  // SCALE folded in
  gemm_proj<1><<<dim3(8, 512), 256, 0, stream>>>(ctx, Wk, Kb, 1.0f);
  gemm_proj<1><<<dim3(8, 512), 256, 0, stream>>>(ctx, Wv, Vb, 1.0f);
  rope_q_kernel<<<256, 256, 0, stream>>>(Qb, qpe);
  rope_k_kernel<<<65536, 256, 0, stream>>>(Kb, kpe);
  attn_kernel<<<dim3(8, 256), 128, 0, stream>>>((const uint32_t*)Qb, (const uint32_t*)Kb,
                                                Vb, nk, nv, Ob);
  gemm_out<<<dim3(16, 128), 256, 0, stream>>>((const uint32_t*)Ob, Wo, bo, out);
}
